// TernaryLSTM_17892833755669
// MI455X (gfx1250) — matrix operand order repaired and checked by reading, not compiled, not run on hardware
//
#include <hip/hip_runtime.h>
#include <hip/hip_bf16.h>
#include <stdint.h>

// Problem dims (from reference): B=64, T=1024, I=512, U=1024, 4U=4096
#define B_  64
#define T_  1024
#define I_  512
#define U_  1024
#define G4  4096

typedef __attribute__((ext_vector_type(16))) __bf16       v16bf;
typedef __attribute__((ext_vector_type(8)))  float        v8f;
typedef __attribute__((ext_vector_type(4)))  unsigned int uv4;

union ABfrag { v16bf v; uv4 q[2]; unsigned int u[8]; };

// round-to-nearest-even f32 -> bf16 (bit trick, no __bf16 arithmetic)
__device__ __forceinline__ unsigned int f32_to_bf16_bits(float f) {
  unsigned int x = __float_as_uint(f);
  x += 0x7fffu + ((x >> 16) & 1u);
  return x >> 16;
}

__device__ __forceinline__ float sigmoidf_(float v) { return 1.0f / (1.0f + __expf(-v)); }
__device__ __forceinline__ float tanhf_(float v)    { return 1.0f - 2.0f / (__expf(2.0f * v) + 1.0f); }

// ---------------------------------------------------------------------------
// Zero LSTM state (h0, c0), the |w| accumulators, and pre-convert x[:,0,:]
// into the first bf16 ping-pong slice (ws is 0xAA-poisoned).
// ---------------------------------------------------------------------------
__global__ void __launch_bounds__(256) init_state(const float* __restrict__ x,
                                                  float* __restrict__ c,
                                                  unsigned short* __restrict__ h,
                                                  unsigned short* __restrict__ xb0,
                                                  float* __restrict__ sums) {
  int i = blockIdx.x * blockDim.x + threadIdx.x;
  if (i < B_ * U_) { c[i] = 0.0f; h[i] = 0; }
  if (i < B_ * I_) {
    int b = i >> 9, xi = i & (I_ - 1);
    xb0[i] = (unsigned short)f32_to_bf16_bits(x[(size_t)b * (T_ * I_) + xi]);
  }
  if (i < 2) sums[i] = 0.0f;
}

// ---------------------------------------------------------------------------
// Pass 1 of ternary quantization: sum |w| (for alpha = mean|w|).
// ---------------------------------------------------------------------------
__global__ void __launch_bounds__(256) abs_sum(const float* __restrict__ w, int n,
                                               float* __restrict__ out) {
  __shared__ float red[256];
  float acc = 0.0f;
  for (int i = blockIdx.x * blockDim.x + threadIdx.x; i < n; i += gridDim.x * blockDim.x)
    acc += fabsf(w[i]);
  red[threadIdx.x] = acc;
  __syncthreads();
  for (int o = 128; o > 0; o >>= 1) {
    if (threadIdx.x < o) red[threadIdx.x] += red[threadIdx.x + o];
    __syncthreads();
  }
  if (threadIdx.x == 0) atomicAdd(out, red[0]);
}

// ---------------------------------------------------------------------------
// Pass 2: ternary quantize (alpha-baked bf16: {-a,0,+a}) AND pre-swizzle into
// the WMMA B-fragment layout for v_wmma_f32_16x16x32_bf16:
//   fragment(nt, kb) = 32x16 tile, stored lane-major, 16 elems/lane:
//     lanes 0-15  : K = kb*32 + (e/8)*16 + e%8     , N = nt*16 + lane
//     lanes 16-31 : K = kb*32 + (e/8)*16 + 8 + e%8 , N = nt*16 + (lane-16)
//   dst index = ((nt*nKb + kb)*32 + lane)*16 + e   (kb minor -> K loop streams)
// Each GEMM lane then loads its fragment as two contiguous b128 loads.
// ---------------------------------------------------------------------------
__global__ void __launch_bounds__(256) quant_pack(const float* __restrict__ w,
                                                  const float* __restrict__ sum,
                                                  int K, int nKb,
                                                  unsigned short* __restrict__ dst) {
  const int n = K * G4;
  const float alpha = *sum / (float)n;
  const float thr = 0.7f * alpha;
  for (int idx = blockIdx.x * blockDim.x + threadIdx.x; idx < n;
       idx += gridDim.x * blockDim.x) {
    int e = idx & 15;
    int lane = (idx >> 4) & 31;
    int frag = idx >> 9;
    int kb = frag % nKb;
    int nt = frag / nKb;
    int k = kb * 32 + ((e >> 3) << 4) + ((lane >> 4) << 3) + (e & 7);
    int col = nt * 16 + (lane & 15);
    float wv = w[(size_t)k * G4 + col];
    unsigned int qv = (fabsf(wv) < thr) ? 0u : f32_to_bf16_bits(copysignf(alpha, wv));
    dst[idx] = (unsigned short)qv;
  }
}

// ---------------------------------------------------------------------------
// One LSTM timestep GEMM: z[64,4096] = x_t[64,512] @ Kq + h[64,1024] @ Rq
// (alpha baked into weights; bias added in the gate kernel). Both A operands
// are pre-converted bf16 row-major buffers -> no conversion VALU on the
// serial critical path; every lane does 4 b128 loads + 1 WMMA per K-block.
// Grid: 128 blocks x 8 waves = 1024 waves, one 16x16 output tile per wave.
// A-fragment layout (16-bit A 16x32): M = lane&15,
//   K(e) = (e/8)*16 + (lane/16)*8 + e%8  -> two contiguous 8-element runs.
// ---------------------------------------------------------------------------
__global__ void __launch_bounds__(256) lstm_gemm_step(
    const unsigned short* __restrict__ xb, const unsigned short* __restrict__ kq,
    const unsigned short* __restrict__ rq, const unsigned short* __restrict__ h,
    float* __restrict__ z) {
  const int lane = threadIdx.x & 31;
  const int wave = threadIdx.x >> 5;
  const int tile = blockIdx.x * 8 + wave;   // 0..1023
  const int mt = tile & 3;                  // 4 M-tiles (batch)
  const int nt = tile >> 2;                 // 256 N-tiles (gate cols)
  const int mrow = mt * 16 + (lane & 15);
  const int khalf = (lane >> 4) << 3;       // 0 or 8

  v8f acc = {};

  // ---- input projection part: K = I_ = 512 (16 blocks of 32) ----
  const unsigned short* xrow = xb + mrow * I_;
  const unsigned short* kqp = kq + ((size_t)nt * 16) * 512 + lane * 16;
  for (int kb = 0; kb < 16; ++kb) {
    ABfrag A, Bm;
    const unsigned short* xp = xrow + kb * 32 + khalf;
    A.q[0] = *reinterpret_cast<const uv4*>(xp);        // K in [khalf, khalf+8)
    A.q[1] = *reinterpret_cast<const uv4*>(xp + 16);   // K in [16+khalf, ...)
    Bm.q[0] = *reinterpret_cast<const uv4*>(kqp);
    Bm.q[1] = *reinterpret_cast<const uv4*>(kqp + 8);
    __builtin_prefetch(kqp + 512, 0, 3);   // near-scope prefetch of next fragment
    acc = __builtin_amdgcn_wmma_f32_16x16x32_bf16(false, A.v, false, Bm.v,
                                                  (short)0, acc, false, false);
    kqp += 512;
  }

  // ---- recurrent part: K = U_ = 1024 (32 blocks of 32) ----
  const unsigned short* hrow = h + mrow * U_;
  const unsigned short* rqp = rq + ((size_t)nt * 32) * 512 + lane * 16;
  for (int kb = 0; kb < 32; ++kb) {
    ABfrag A, Bm;
    const unsigned short* hp = hrow + kb * 32 + khalf;
    A.q[0] = *reinterpret_cast<const uv4*>(hp);
    A.q[1] = *reinterpret_cast<const uv4*>(hp + 16);
    Bm.q[0] = *reinterpret_cast<const uv4*>(rqp);
    Bm.q[1] = *reinterpret_cast<const uv4*>(rqp + 8);
    __builtin_prefetch(rqp + 512, 0, 3);
    acc = __builtin_amdgcn_wmma_f32_16x16x32_bf16(false, A.v, false, Bm.v,
                                                  (short)0, acc, false, false);
    rqp += 512;
  }

  // C/D layout: VGPR r -> M = (lane/16)*8 + r, N = lane&15
  const int col = nt * 16 + (lane & 15);
  float* zp = z + (size_t)(mt * 16 + khalf) * G4 + col;
#pragma unroll
  for (int r = 0; r < 8; ++r) zp[(size_t)r * G4] = acc[r];
}

// ---------------------------------------------------------------------------
// Gates + state update + output (keras gate order i, f, g, o).
// Writes h as bf16 (next step's WMMA A operand) and f32 to d_out.
// Also pre-converts next timestep's x slice into the ping-pong bf16 buffer,
// keeping conversion VALU off the GEMM critical path.
// ---------------------------------------------------------------------------
__global__ void __launch_bounds__(256) lstm_gate_step(
    const float* __restrict__ z, const float* __restrict__ bias,
    float* __restrict__ c, unsigned short* __restrict__ h,
    const float* __restrict__ x, unsigned short* __restrict__ xnext,
    float* __restrict__ out, int t) {
  int idx = blockIdx.x * blockDim.x + threadIdx.x;  // 0..65535
  int b = idx >> 10;
  int u = idx & (U_ - 1);
  const float* zr = z + (size_t)b * G4;
  float zi = zr[u]          + bias[u];
  float zf = zr[U_ + u]     + bias[U_ + u];
  float zg = zr[2 * U_ + u] + bias[2 * U_ + u];
  float zo = zr[3 * U_ + u] + bias[3 * U_ + u];
  float ig = sigmoidf_(zi);
  float fg = sigmoidf_(zf);
  float gg = tanhf_(zg);
  float og = sigmoidf_(zo);
  float cn = fg * c[idx] + ig * gg;
  c[idx] = cn;
  float hn = og * tanhf_(cn);
  h[idx] = (unsigned short)f32_to_bf16_bits(hn);
  out[(size_t)b * (T_ * U_) + (size_t)t * U_ + u] = hn;

  // stage x[:, t+1, :] as bf16 for the next step's GEMM
  if ((t + 1) < T_ && idx < B_ * I_) {
    int xb_ = idx >> 9, xi = idx & (I_ - 1);
    xnext[idx] = (unsigned short)f32_to_bf16_bits(
        x[(size_t)xb_ * (T_ * I_) + (size_t)(t + 1) * I_ + xi]);
  }
}

// ---------------------------------------------------------------------------
// Workspace layout (~13.5 MiB; weights stay L2-resident across all 1024 steps):
//   [0, 4MiB)              kq packed bf16  (512  x 4096)
//   [4MiB, 12MiB)          rq packed bf16  (1024 x 4096)
//   [12MiB, +128KiB)       h   bf16        (64 x 1024)
//   [.., +256KiB)          c   f32         (64 x 1024)
//   [.., +1MiB)            z   f32         (64 x 4096)
//   [.., +64KiB)           xb0 bf16        (64 x 512)  ping
//   [.., +64KiB)           xb1 bf16        (64 x 512)  pong
//   [.., +8B)              |w| sums (kernel, recurrent)
// ---------------------------------------------------------------------------
extern "C" void kernel_launch(void* const* d_in, const int* in_sizes, int n_in,
                              void* d_out, int out_size, void* d_ws, size_t ws_size,
                              hipStream_t stream) {
  (void)in_sizes; (void)n_in; (void)out_size; (void)ws_size;
  const float* x    = (const float*)d_in[0];
  const float* ker  = (const float*)d_in[1];
  const float* rker = (const float*)d_in[2];
  const float* bias = (const float*)d_in[3];
  float* out = (float*)d_out;

  uint8_t* ws = (uint8_t*)d_ws;
  unsigned short* kqP = (unsigned short*)(ws);
  unsigned short* rqP = (unsigned short*)(ws + (4u << 20));
  unsigned short* hB  = (unsigned short*)(ws + (12u << 20));
  float* cB   = (float*)(ws + (12u << 20) + (128u << 10));
  float* zB   = (float*)(ws + (12u << 20) + (384u << 10));
  unsigned short* xbP[2];
  xbP[0] = (unsigned short*)(ws + (13u << 20) + (384u << 10));
  xbP[1] = (unsigned short*)(ws + (13u << 20) + (448u << 10));
  float* sums = (float*)(ws + (13u << 20) + (512u << 10));

  init_state<<<256, 256, 0, stream>>>(x, cB, hB, xbP[0], sums);
  abs_sum<<<256, 256, 0, stream>>>(ker,  I_ * G4, sums + 0);
  abs_sum<<<256, 256, 0, stream>>>(rker, U_ * G4, sums + 1);
  quant_pack<<<512, 256, 0, stream>>>(ker,  sums + 0, I_, 16, kqP);
  quant_pack<<<1024, 256, 0, stream>>>(rker, sums + 1, U_, 32, rqP);

  for (int t = 0; t < T_; ++t) {
    lstm_gemm_step<<<128, 256, 0, stream>>>(xbP[t & 1], kqP, rqP, hB, zB);
    lstm_gate_step<<<256, 256, 0, stream>>>(zB, bias, cB, hB, x, xbP[(t + 1) & 1],
                                            out, t);
  }
}
